// NNUE_14723147891125
// MI455X (gfx1250) — matrix operand order repaired and checked by reading, not compiled, not run on hardware
//
#include <hip/hip_runtime.h>
#include <hip/hip_bf16.h>

#define FEAT 41024
#define HID  256
#define BATCH 2048

typedef __attribute__((ext_vector_type(16))) __bf16 v16bf;
typedef __attribute__((ext_vector_type(8)))  float  v8f;
typedef __attribute__((ext_vector_type(4)))  float  v4f;

// ---------------------------------------------------------------------------
// Prep: W_in [HID, FEAT] f32 (row-major) -> Wt2 bf16 in WMMA-B fragment-ready
// layout: Wt2[kb][n16][kl][i] with kb=K/32 chunk, n16=N/16 chunk, kl=B-fragment lane
// (N%16 = kl%16), i=element with K%32 = i%8 + 8*(kl/16) + 16*(i/8). Each 1KB block is exactly one
// wave's v16bf B-fragment load, 32B/lane fully coalesced.
// ---------------------------------------------------------------------------
__global__ __launch_bounds__(256)
void nnue_prep_w(const float* __restrict__ W_in, __bf16* __restrict__ Wt2) {
    const int kb = blockIdx.x;   // 0..FEAT/32-1
    const int nt = blockIdx.y;   // 0..HID/16-1
    __shared__ float tile[16 * 33];  // padded: no bank conflicts on transpose
    for (int e = threadIdx.x; e < 512; e += 256) {
        const int nn = e >> 5, kk = e & 31;  // coalesced 128B reads per n-row
        tile[nn * 33 + kk] = W_in[(size_t)(nt * 16 + nn) * FEAT + kb * 32 + kk];
    }
    __syncthreads();
    __bf16* out = Wt2 + ((size_t)kb * (HID / 16) + nt) * 512;
    for (int e = threadIdx.x; e < 512; e += 256) {
        const int kl = e >> 4, i = e & 15;
        out[e] = (__bf16)tile[(kl & 15) * 33 + (i & 7) + 8 * (kl >> 4) + 16 * (i >> 3)];  // coalesced 1KB block write
    }
}

// Build one 16x32 bf16 A-fragment from 16 fp32 feature values (NT stream).
// Per ISA layout: lane<16 -> row M=lane, K={0..7,16..23}; lane>=16 -> same
// row, K={8..15,24..31}; element 2v+h of the fragment = VGPR v half h.
__device__ __forceinline__ v16bf load_a_frag(const float* a0) {
    v4f x0 = __builtin_nontemporal_load((const v4f*)(a0));
    v4f x1 = __builtin_nontemporal_load((const v4f*)(a0 + 4));
    v4f x2 = __builtin_nontemporal_load((const v4f*)(a0 + 16));
    v4f x3 = __builtin_nontemporal_load((const v4f*)(a0 + 20));
    v16bf a;
#pragma unroll
    for (int i = 0; i < 4; ++i) {
        a[i]      = (__bf16)x0[i];
        a[4 + i]  = (__bf16)x1[i];
        a[8 + i]  = (__bf16)x2[i];
        a[12 + i] = (__bf16)x3[i];
    }
    return a;
}

// ---------------------------------------------------------------------------
// Accumulator GEMM: H[:, c*256 .. c*256+255] = relu(feat @ W_in^T + b_in)
// Block = 256 threads (8 waves, wave32). Block tile 32(M) x 256(N); wave tile
// 16 x 64 => 4 accumulators of v_wmma_f32_16x16x32_bf16. K-loop strides by 2
// chunks with two live A fragments so the bf16-WMMA WAR hazard distance (4
// co-exec slots) is met structurally, keeping accumulators pinned in place.
// ---------------------------------------------------------------------------
__global__ __launch_bounds__(256)
void nnue_accum_gemm(const float* __restrict__ whiteF,
                     const float* __restrict__ blackF,
                     const __bf16* __restrict__ Wt2,
                     const float* __restrict__ b_in,
                     float* __restrict__ H) {
    const int lane   = threadIdx.x & 31;
    const int wave   = threadIdx.x >> 5;   // 0..7
    const int wave_m = wave >> 2;          // 0..1
    const int wave_n = wave & 3;           // 0..3
    const int c      = blockIdx.y;         // 0 = white, 1 = black
    const float* feat = c ? blackF : whiteF;

    const int m0  = blockIdx.x * 32 + wave_m * 16;
    const int nl  = lane & 15;
    const int sel = lane >> 4;

    const int row = m0 + nl;
    const float* pA = feat + (size_t)row * FEAT + sel * 8;
    // B-fragment base for this wave: n-blocks wave_n*4 .. wave_n*4+3, lane -> N=lane%16, K set by lane/16
    const __bf16* pB = Wt2 + (size_t)(wave_n * 4) * 512 + lane * 16;

    float bias[4];
#pragma unroll
    for (int j = 0; j < 4; ++j) bias[j] = b_in[wave_n * 64 + j * 16 + nl];

    v8f acc[4];
#pragma unroll
    for (int j = 0; j < 4; ++j) acc[j] = (v8f)(0.0f);

    for (int kb = 0; kb < FEAT / 32; kb += 2) {   // 641 exact iterations
        const float* a_ptr = pA + (size_t)kb * 32;
        v16bf a0 = load_a_frag(a_ptr);        // chunk kb
        v16bf a1 = load_a_frag(a_ptr + 32);   // chunk kb+1
        __builtin_prefetch((const void*)(a_ptr + 32 * 32), 0, 0);  // global_prefetch_b8

        const __bf16* b0 = pB + (size_t)kb * ((HID / 16) * 512);
#pragma unroll
        for (int j = 0; j < 4; ++j) {
            v16bf b = *(const v16bf*)(b0 + j * 512);  // L2-resident weights
            acc[j] = __builtin_amdgcn_wmma_f32_16x16x32_bf16(
                false, a0, false, b, (short)0, acc[j], false, false);
        }
        const __bf16* b1 = b0 + (HID / 16) * 512;
#pragma unroll
        for (int j = 0; j < 4; ++j) {
            v16bf b = *(const v16bf*)(b1 + j * 512);
            acc[j] = __builtin_amdgcn_wmma_f32_16x16x32_bf16(
                false, a1, false, b, (short)0, acc[j], false, false);
        }
    }

    // Epilogue: bias + ReLU, write combined hidden [BATCH, 2*HID] fp32.
    // C/D layout: VGPR r, lanes 0-15 -> M=r, N=lane; lanes 16-31 -> M=8+r.
#pragma unroll
    for (int j = 0; j < 4; ++j) {
        const int n = wave_n * 64 + j * 16 + nl;
#pragma unroll
        for (int r = 0; r < 8; ++r) {
            const int m = m0 + r + 8 * sel;
            const float v = acc[j][r] + bias[j];
            H[(size_t)m * (2 * HID) + c * HID + n] = fmaxf(v, 0.0f);
        }
    }
}

// ---------------------------------------------------------------------------
// Tail: x1 = relu(H @ W_h1^T + b_h1); x2 = relu(x1 @ W_h2^T + b_h2);
//       out = x2 @ W_out^T + b_out.   One wave per batch row, lane = unit id.
// ~70 MFLOP total: VALU fp32 is ample; wave32 shuffles do the row exchanges.
// ---------------------------------------------------------------------------
__global__ __launch_bounds__(256)
void nnue_tail(const float* __restrict__ H,
               const float* __restrict__ W_h1, const float* __restrict__ b_h1,
               const float* __restrict__ W_h2, const float* __restrict__ b_h2,
               const float* __restrict__ W_out, const float* __restrict__ b_out,
               float* __restrict__ out) {
    const int lane = threadIdx.x & 31;
    const int wave = threadIdx.x >> 5;
    const int r    = blockIdx.x * 8 + wave;

    const float* hrow = H + (size_t)r * (2 * HID);
    const float* w1   = W_h1 + (size_t)lane * (2 * HID);
    float acc = b_h1[lane];
    for (int j = 0; j < 2 * HID; j += 4) {
        v4f h = *(const v4f*)(hrow + j);   // broadcast across lanes
        v4f w = *(const v4f*)(w1 + j);
        acc += h[0] * w[0] + h[1] * w[1] + h[2] * w[2] + h[3] * w[3];
    }
    const float x1 = fmaxf(acc, 0.0f);

    float acc2 = b_h2[lane];
#pragma unroll
    for (int i = 0; i < 32; ++i)
        acc2 += __shfl(x1, i, 32) * W_h2[lane * 32 + i];
    const float x2 = fmaxf(acc2, 0.0f);

    float p = x2 * W_out[lane];
#pragma unroll
    for (int off = 16; off > 0; off >>= 1) p += __shfl_down(p, off, 32);
    if (lane == 0) out[r] = p + b_out[0];
}

// ---------------------------------------------------------------------------
// Workspace layout:
//   [0, 21004288)            Wt2  : FEAT*HID bf16 (fragment-ready W_in)
//   [21004288, 25198592)     H    : BATCH x 2*HID fp32 combined hidden
// ---------------------------------------------------------------------------
extern "C" void kernel_launch(void* const* d_in, const int* in_sizes, int n_in,
                              void* d_out, int out_size, void* d_ws, size_t ws_size,
                              hipStream_t stream) {
    const float* whiteF = (const float*)d_in[0];
    const float* blackF = (const float*)d_in[1];
    const float* W_in   = (const float*)d_in[2];
    const float* b_in   = (const float*)d_in[3];
    const float* W_h1   = (const float*)d_in[4];
    const float* b_h1   = (const float*)d_in[5];
    const float* W_h2   = (const float*)d_in[6];
    const float* b_h2   = (const float*)d_in[7];
    const float* W_out  = (const float*)d_in[8];
    const float* b_out  = (const float*)d_in[9];
    float* out = (float*)d_out;

    __bf16* Wt2 = (__bf16*)d_ws;
    float*  H   = (float*)((char*)d_ws + (size_t)FEAT * HID * sizeof(__bf16));

    nnue_prep_w<<<dim3(FEAT / 32, HID / 16), 256, 0, stream>>>(W_in, Wt2);
    nnue_accum_gemm<<<dim3(BATCH / 32, 2), 256, 0, stream>>>(whiteF, blackF, Wt2, b_in, H);
    nnue_tail<<<dim3(BATCH / 8), 256, 0, stream>>>(H, W_h1, b_h1, W_h2, b_h2, W_out, b_out, out);
}
